// S4Model_23613730193720
// MI455X (gfx1250) — matrix operand order repaired and checked by reading, not compiled, not run on hardware
//
#include <hip/hip_runtime.h>
#include <math.h>

// ---------------- constants ----------------
constexpr int HDIM = 768;
constexpr int NST  = 16;
constexpr int NL   = 6;
constexpr int BB   = 8;
constexpr int LMAX = 2048;
constexpr int NOUT = 38;
constexpr int NOUTP = 48;                 // padded to 3 N-tiles
constexpr int PCH  = 16;                  // scan chunks per sequence
constexpr size_t XY_ELEMS = (size_t)BB * LMAX * HDIM;   // 12,582,912
constexpr size_t SBUF_ELEMS = (size_t)BB * HDIM * PCH * NST * 2; // 3,145,728

typedef __bf16 bf16;
typedef __attribute__((ext_vector_type(16))) __bf16 v16bf;
typedef __attribute__((ext_vector_type(8)))  float  v8f;
typedef __attribute__((ext_vector_type(4)))  unsigned int u32x4;

union FragBF { u32x4 q[2]; v16bf v; };

// async global->LDS staging of one 16B chunk (tracked by ASYNCcnt)
__device__ inline void async_copy_b128(const bf16* src, bf16* dst_lds) {
  unsigned int  laddr = (unsigned int)(unsigned long long)(uintptr_t)dst_lds; // LDS byte addr
  unsigned long long gaddr = (unsigned long long)(uintptr_t)src;
  asm volatile("global_load_async_to_lds_b128 %0, %1, off"
               :: "v"(laddr), "v"(gaddr) : "memory");
}
__device__ inline void async_wait0() {
  asm volatile("s_wait_asynccnt 0x0" ::: "memory");
}

// ---------------- weight prep ----------------
__global__ void k_transpose_bf16(const float* __restrict__ src, bf16* __restrict__ dst,
                                 int R, int C) {
  int tid = blockIdx.x * blockDim.x + threadIdx.x;
  if (tid >= R * C) return;
  int c = tid / R, r = tid - c * R;
  dst[tid] = (bf16)src[r * C + c];
}

__global__ void k_convert_bf16(const float* __restrict__ src, bf16* __restrict__ dst, int n) {
  int tid = blockIdx.x * blockDim.x + threadIdx.x;
  if (tid < n) dst[tid] = (bf16)src[tid];
}

__global__ void k_build_wout(const float* __restrict__ src /*768 x 38*/,
                             bf16* __restrict__ dst /*48 x 768*/) {
  int tid = blockIdx.x * blockDim.x + threadIdx.x;
  if (tid >= NOUTP * HDIM) return;
  int n = tid / HDIM, h = tid - n * HDIM;
  dst[tid] = (n < NOUT) ? (bf16)src[h * NOUT + n] : (bf16)0.0f;
}

// ---------------- encoder stage 1 ----------------
__global__ void k_encoder1(const float* __restrict__ xraw, const float* __restrict__ w1,
                           const float* __restrict__ b1, bf16* __restrict__ Abf, int M) {
  int tid = blockIdx.x * blockDim.x + threadIdx.x;
  if (tid >= M * HDIM) return;
  int t = tid / HDIM, j = tid - t * HDIM;
  float s = b1[j];
#pragma unroll
  for (int k = 0; k < 8; ++k) s += xraw[t * 8 + k] * w1[k * HDIM + j];
  Abf[tid] = (bf16)(s / (1.0f + fabsf(s)));   // soft_sign
}

// ---------------- WMMA GEMM (2 M-tiles/wave): C = A @ B^T + bias ----------------
__global__ __launch_bounds__(128) void k_gemm_bias(
    const bf16* __restrict__ A, const bf16* __restrict__ Bt,
    const float* __restrict__ bias, float* __restrict__ C,
    int K, int ldc, int tiles_m2, int tiles_n, int Nvalid) {
  int wave = blockIdx.x * 4 + (threadIdx.x >> 5);
  int tile_m2 = wave % tiles_m2;
  int tile_n  = wave / tiles_m2;
  if (tile_n >= tiles_n) return;
  int lane = threadIdx.x & 31;
  int row  = lane & 15;
  int ka   = (lane >> 4) << 3;    // A-frag K half offset: 0 or 8
  int kb   = (lane >> 4) << 3;    // B-frag K half offset: 0 or 8
  int m0   = tile_m2 * 32;
  const bf16* a0_ptr = A  + (size_t)(m0 + row) * K + ka;
  const bf16* a1_ptr = a0_ptr + (size_t)16 * K;
  const bf16* b_ptr  = Bt + (size_t)(tile_n * 16 + row) * K + kb;
  v8f acc0 = {}, acc1 = {};
  for (int k0 = 0; k0 < K; k0 += 32) {
    FragBF fa0, fa1, fb;
    const u32x4* pa0 = (const u32x4*)(a0_ptr + k0);
    const u32x4* pa1 = (const u32x4*)(a1_ptr + k0);
    const u32x4* pb  = (const u32x4*)(b_ptr + k0);
    fa0.q[0] = pa0[0]; fa0.q[1] = pa0[2];   // K: ka..ka+7, ka+16..ka+23
    fa1.q[0] = pa1[0]; fa1.q[1] = pa1[2];
    fb.q[0]  = pb[0];  fb.q[1]  = pb[2];    // K: kb..kb+7, kb+16..kb+23
    acc0 = __builtin_amdgcn_wmma_f32_16x16x32_bf16(false, fa0.v, false, fb.v,
                                                   (short)0, acc0, false, false);
    acc1 = __builtin_amdgcn_wmma_f32_16x16x32_bf16(false, fa1.v, false, fb.v,
                                                   (short)0, acc1, false, false);
  }
  int col = lane & 15;
  int d   = tile_n * 16 + col;
  int rb  = m0 + ((lane >> 4) << 3);
  if (d < Nvalid) {
    float bv = bias[d];
#pragma unroll
    for (int i = 0; i < 8; ++i) {
      C[(size_t)(rb + i) * ldc + d]      = acc0[i] + bv;
      C[(size_t)(rb + 16 + i) * ldc + d] = acc1[i] + bv;
    }
  }
}

// ---------------- WMMA GEMM + GLU + residual, LDS-staged weights ----------------
// Block = 128 threads = 4 waves; covers 128 M-rows x 16 N-cols; K = HDIM.
// Weight tiles (a-rows d..d+15 and g-rows d+H..) staged once into LDS via
// global_load_async_to_lds_b128 (ASYNCcnt), shared by all 4 waves.
__global__ __launch_bounds__(128) void k_gemm_glu(
    const bf16* __restrict__ A, const bf16* __restrict__ Wt /*(2H)xK*/,
    const float* __restrict__ bias2h, const float* __restrict__ U,
    float* __restrict__ Y, int K, int Lk, int ds, int Lfull,
    int tiles_m8, int tiles_n) {
  __shared__ bf16 smB[32 * HDIM];                 // 48 KB: rows 0-15 = a, 16-31 = g
  int bm = blockIdx.x % tiles_m8;
  int bn = blockIdx.x / tiles_m8;

  // ---- stage B tiles: 32 rows x 768 cols = 3072 x 16B chunks, 24 per thread ----
  {
    const bf16* wa = Wt + (size_t)(bn * 16) * K;           // a rows
    const bf16* wg = Wt + (size_t)(HDIM + bn * 16) * K;    // g rows
    for (int u = threadIdx.x; u < 32 * 96; u += 128) {     // 96 chunks per row
      int rowu = u / 96;
      int colu = (u - rowu * 96) * 8;                      // element offset
      const bf16* src = (rowu < 16) ? (wa + (size_t)rowu * K + colu)
                                    : (wg + (size_t)(rowu - 16) * K + colu);
      async_copy_b128(src, &smB[rowu * HDIM + colu]);
    }
    async_wait0();
    __syncthreads();
  }

  int wv   = threadIdx.x >> 5;
  int lane = threadIdx.x & 31;
  int row  = lane & 15;
  int ka   = (lane >> 4) << 3;
  int kb   = (lane >> 4) << 3;
  int m0   = bm * 128 + wv * 32;
  const bf16* a0_ptr = A + (size_t)(m0 + row) * K + ka;
  const bf16* a1_ptr = a0_ptr + (size_t)16 * K;
  const bf16* sba = &smB[(lane & 15) * HDIM + kb];
  const bf16* sbg = &smB[(16 + (lane & 15)) * HDIM + kb];
  v8f acc0a = {}, acc0g = {}, acc1a = {}, acc1g = {};
  for (int k0 = 0; k0 < K; k0 += 32) {
    FragBF fa0, fa1, fba, fbg;
    const u32x4* pa0 = (const u32x4*)(a0_ptr + k0);
    const u32x4* pa1 = (const u32x4*)(a1_ptr + k0);
    const u32x4* pba = (const u32x4*)(sba + k0);
    const u32x4* pbg = (const u32x4*)(sbg + k0);
    fa0.q[0] = pa0[0]; fa0.q[1] = pa0[2];
    fa1.q[0] = pa1[0]; fa1.q[1] = pa1[2];
    fba.q[0] = pba[0]; fba.q[1] = pba[2];
    fbg.q[0] = pbg[0]; fbg.q[1] = pbg[2];
    acc0a = __builtin_amdgcn_wmma_f32_16x16x32_bf16(false, fa0.v, false, fba.v,
                                                    (short)0, acc0a, false, false);
    acc0g = __builtin_amdgcn_wmma_f32_16x16x32_bf16(false, fa0.v, false, fbg.v,
                                                    (short)0, acc0g, false, false);
    acc1a = __builtin_amdgcn_wmma_f32_16x16x32_bf16(false, fa1.v, false, fba.v,
                                                    (short)0, acc1a, false, false);
    acc1g = __builtin_amdgcn_wmma_f32_16x16x32_bf16(false, fa1.v, false, fbg.v,
                                                    (short)0, acc1g, false, false);
  }
  int col = lane & 15;
  int d   = bn * 16 + col;
  float badd = bias2h[d], bg = bias2h[d + HDIM];
#pragma unroll
  for (int mi = 0; mi < 2; ++mi) {
    int rb = m0 + mi * 16 + ((lane >> 4) << 3);
    v8f aa = mi ? acc1a : acc0a;
    v8f gg = mi ? acc1g : acc0g;
#pragma unroll
    for (int i = 0; i < 8; ++i) {
      int t  = rb + i;
      int b  = t / Lk;
      int lk = t - b * Lk;
      int urow = b * Lfull + lk * ds;       // residual u at the kept timestep
      float av = aa[i] + badd;
      float gv = gg[i] + bg;
      float z  = av * (1.0f / (1.0f + expf(-gv)));
      Y[(size_t)t * HDIM + d] = z + U[(size_t)urow * HDIM + d];
    }
  }
}

// ---------------- S4 discretization helper ----------------
__device__ inline void s4_params(const float* __restrict__ log_dt,
                                 const float* __restrict__ log_A_real,
                                 const float* __restrict__ A_imag, int h,
                                 float* er, float* ei, float* dbr, float* dbi) {
  float dt = expf(log_dt[h]);
#pragma unroll
  for (int n = 0; n < NST; ++n) {
    float Ar = -expf(log_A_real[h * NST + n]);
    float Ai = A_imag[h * NST + n];
    float mg = expf(dt * Ar);
    float eAr = mg * cosf(dt * Ai), eAi = mg * sinf(dt * Ai);
    float den = Ar * Ar + Ai * Ai;
    float nr = eAr - 1.0f, ni = eAi;
    er[n] = eAr; ei[n] = eAi;
    dbr[n] = (nr * Ar + ni * Ai) / den;
    dbi[n] = (ni * Ar - nr * Ai) / den;
  }
}

// ---------------- scan phase 1: chunk-local final states ----------------
__global__ __launch_bounds__(256) void k_scan_p1(
    const float* __restrict__ X, float* __restrict__ SBuf,
    const float* __restrict__ log_dt, const float* __restrict__ log_A_real,
    const float* __restrict__ A_imag, int L, int dir) {
  int tid = blockIdx.x * blockDim.x + threadIdx.x;
  if (tid >= BB * PCH * HDIM) return;
  int b = tid / (PCH * HDIM);
  int r = tid - b * (PCH * HDIM);
  int c = r / HDIM, h = r - c * HDIM;
  float er[NST], ei[NST], dbr[NST], dbi[NST];
  s4_params(log_dt, log_A_real, A_imag, h, er, ei, dbr, dbi);
  float sr[NST] = {}, si[NST] = {};
  int CL = L / PCH;
  const float* xp = X + ((size_t)b * L + (size_t)c * CL) * HDIM + h;
  if (dir == 0) {
    for (int j = 0; j < CL; ++j) {
      __builtin_prefetch(xp + (size_t)(j + 8) * HDIM, 0, 1);
      float u = xp[(size_t)j * HDIM];
#pragma unroll
      for (int n = 0; n < NST; ++n) {
        float s_r = er[n] * sr[n] - ei[n] * si[n] + dbr[n] * u;
        float s_i = er[n] * si[n] + ei[n] * sr[n] + dbi[n] * u;
        sr[n] = s_r; si[n] = s_i;
      }
    }
  } else {
    for (int j = CL - 1; j >= 0; --j) {
      __builtin_prefetch(xp + (size_t)(j - 8) * HDIM, 0, 1);
      float u = xp[(size_t)j * HDIM];
#pragma unroll
      for (int n = 0; n < NST; ++n) {
        float s_r = er[n] * sr[n] - ei[n] * si[n] + dbr[n] * u;
        float s_i = er[n] * si[n] + ei[n] * sr[n] + dbi[n] * u;
        sr[n] = s_r; si[n] = s_i;
      }
    }
  }
  float* sp = SBuf + (((size_t)b * HDIM + h) * PCH + c) * (NST * 2);
#pragma unroll
  for (int n = 0; n < NST; ++n) { sp[2 * n] = sr[n]; sp[2 * n + 1] = si[n]; }
}

// ---------------- scan phase 2: combine chunk states (prefix) ----------------
__global__ __launch_bounds__(256) void k_scan_combine(
    float* __restrict__ SBuf, const float* __restrict__ log_dt,
    const float* __restrict__ log_A_real, const float* __restrict__ A_imag,
    int L, int dir) {
  int tid = blockIdx.x * blockDim.x + threadIdx.x;
  if (tid >= BB * HDIM * NST) return;
  int bh = tid / NST, n = tid - bh * NST;
  int h = bh % HDIM;
  float dt = expf(log_dt[h]);
  float Ar = -expf(log_A_real[h * NST + n]);
  float Ai = A_imag[h * NST + n];
  float CLf = (float)(L / PCH);
  float mg = expf(dt * Ar * CLf);                   // eA^CL
  float eR = mg * cosf(dt * Ai * CLf);
  float eI = mg * sinf(dt * Ai * CLf);
  float* base = SBuf + (size_t)bh * PCH * (NST * 2) + 2 * n;
  float pr = 0.0f, pi = 0.0f;
  if (dir == 0) {
    for (int c = 0; c < PCH; ++c) {
      float* sp = base + (size_t)c * (NST * 2);
      float fr = sp[0], fi = sp[1];
      sp[0] = pr; sp[1] = pi;                       // s_init[c]
      float npr = eR * pr - eI * pi + fr;
      float npi = eR * pi + eI * pr + fi;
      pr = npr; pi = npi;
    }
  } else {
    for (int c = PCH - 1; c >= 0; --c) {
      float* sp = base + (size_t)c * (NST * 2);
      float fr = sp[0], fi = sp[1];
      sp[0] = pr; sp[1] = pi;                       // r_init[c]
      float npr = eR * pr - eI * pi + fr;
      float npi = eR * pi + eI * pr + fi;
      pr = npr; pi = npi;
    }
  }
}

// ---------------- scan phase 3 fwd: emit yacc = D*u + 2Re(C0*s) ----------------
__global__ __launch_bounds__(256) void k_scan_p3_fwd(
    const float* __restrict__ X, float* __restrict__ Y,
    const float* __restrict__ SBuf,
    const float* __restrict__ log_dt, const float* __restrict__ log_A_real,
    const float* __restrict__ A_imag, const float* __restrict__ C0,
    const float* __restrict__ Dv, int L) {
  int tid = blockIdx.x * blockDim.x + threadIdx.x;
  if (tid >= BB * PCH * HDIM) return;
  int b = tid / (PCH * HDIM);
  int r = tid - b * (PCH * HDIM);
  int c = r / HDIM, h = r - c * HDIM;
  float er[NST], ei[NST], dbr[NST], dbi[NST], cr[NST], ci[NST];
  s4_params(log_dt, log_A_real, A_imag, h, er, ei, dbr, dbi);
#pragma unroll
  for (int n = 0; n < NST; ++n) {
    cr[n] = 2.0f * C0[(h * NST + n) * 2 + 0];
    ci[n] = 2.0f * C0[(h * NST + n) * 2 + 1];
  }
  float Dval = Dv[h];
  float sr[NST], si[NST];
  const float* sp = SBuf + (((size_t)b * HDIM + h) * PCH + c) * (NST * 2);
#pragma unroll
  for (int n = 0; n < NST; ++n) { sr[n] = sp[2 * n]; si[n] = sp[2 * n + 1]; }
  int CL = L / PCH;
  const float* xp = X + ((size_t)b * L + (size_t)c * CL) * HDIM + h;
  float*       yp = Y + ((size_t)b * L + (size_t)c * CL) * HDIM + h;
  for (int j = 0; j < CL; ++j) {
    __builtin_prefetch(xp + (size_t)(j + 8) * HDIM, 0, 1);
    float u = xp[(size_t)j * HDIM];
    float acc = Dval * u;
#pragma unroll
    for (int n = 0; n < NST; ++n) {
      float s_r = er[n] * sr[n] - ei[n] * si[n] + dbr[n] * u;
      float s_i = er[n] * si[n] + ei[n] * sr[n] + dbi[n] * u;
      sr[n] = s_r; si[n] = s_i;
      acc += cr[n] * s_r - ci[n] * s_i;
    }
    yp[(size_t)j * HDIM] = acc;
  }
}

// ---------- scan phase 3 bwd: y += 2Re(C1*r); exact GELU; compact kept tokens ----------
__global__ __launch_bounds__(256) void k_scan_p3_bwd(
    const float* __restrict__ X, const float* __restrict__ Y, bf16* __restrict__ Abf,
    const float* __restrict__ SBuf,
    const float* __restrict__ log_dt, const float* __restrict__ log_A_real,
    const float* __restrict__ A_imag, const float* __restrict__ C1,
    int L, int ds) {
  int tid = blockIdx.x * blockDim.x + threadIdx.x;
  if (tid >= BB * PCH * HDIM) return;
  int b = tid / (PCH * HDIM);
  int r = tid - b * (PCH * HDIM);
  int c = r / HDIM, h = r - c * HDIM;
  float er[NST], ei[NST], dbr[NST], dbi[NST], cr[NST], ci[NST];
  s4_params(log_dt, log_A_real, A_imag, h, er, ei, dbr, dbi);
#pragma unroll
  for (int n = 0; n < NST; ++n) {
    cr[n] = 2.0f * C1[(h * NST + n) * 2 + 0];
    ci[n] = 2.0f * C1[(h * NST + n) * 2 + 1];
  }
  float rr[NST], ri[NST];
  const float* sp = SBuf + (((size_t)b * HDIM + h) * PCH + c) * (NST * 2);
#pragma unroll
  for (int n = 0; n < NST; ++n) { rr[n] = sp[2 * n]; ri[n] = sp[2 * n + 1]; }
  int CL = L / PCH;
  int Lk = L / ds;
  int l0 = c * CL;
  const float* xp = X + ((size_t)b * L + l0) * HDIM + h;
  const float* yp = Y + ((size_t)b * L + l0) * HDIM + h;
  for (int j = CL - 1; j >= 0; --j) {
    __builtin_prefetch(xp + (size_t)(j - 8) * HDIM, 0, 1);
    float ysum = yp[(size_t)j * HDIM];          // y_fwd + D*u from fwd pass
#pragma unroll
    for (int n = 0; n < NST; ++n) ysum += cr[n] * rr[n] - ci[n] * ri[n];
    int l = l0 + j;
    if (ds == 1 || (l & 1) == 0) {
      int t = b * Lk + l / ds;
      float g = 0.5f * ysum * (1.0f + erff(ysum * 0.70710678118f));  // exact GELU
      Abf[(size_t)t * HDIM + h] = (bf16)g;
    }
    float u = xp[(size_t)j * HDIM];
#pragma unroll
    for (int n = 0; n < NST; ++n) {
      float r_r = er[n] * rr[n] - ei[n] * ri[n] + dbr[n] * u;
      float r_i = er[n] * ri[n] + ei[n] * rr[n] + dbi[n] * u;
      rr[n] = r_r; ri[n] = r_i;
    }
  }
}

// ---------------- layernorm (per token over H=768) ----------------
__global__ __launch_bounds__(256) void k_layernorm(
    const float* __restrict__ Xr, float* __restrict__ Xo,
    const float* __restrict__ w, const float* __restrict__ bn) {
  int t = blockIdx.x;
  int tx = threadIdx.x;
  __shared__ float red[16];
  const float* p = Xr + (size_t)t * HDIM;
  float v0 = p[tx], v1 = p[tx + 256], v2 = p[tx + 512];
  float s = v0 + v1 + v2;
  float sq = v0 * v0 + v1 * v1 + v2 * v2;
#pragma unroll
  for (int off = 16; off > 0; off >>= 1) {
    s  += __shfl_xor(s,  off, 32);
    sq += __shfl_xor(sq, off, 32);
  }
  if ((tx & 31) == 0) { red[tx >> 5] = s; red[8 + (tx >> 5)] = sq; }
  __syncthreads();
  float S = 0.0f, SQ = 0.0f;
#pragma unroll
  for (int i = 0; i < 8; ++i) { S += red[i]; SQ += red[8 + i]; }
  float mu = S * (1.0f / HDIM);
  float var = SQ * (1.0f / HDIM) - mu * mu;
  float rstd = rsqrtf(var + 1e-5f);
  float* o = Xo + (size_t)t * HDIM;
  o[tx]       = (v0 - mu) * rstd * w[tx]       + bn[tx];
  o[tx + 256] = (v1 - mu) * rstd * w[tx + 256] + bn[tx + 256];
  o[tx + 512] = (v2 - mu) * rstd * w[tx + 512] + bn[tx + 512];
}

// ---------------- launch ----------------
extern "C" void kernel_launch(void* const* d_in, const int* in_sizes, int n_in,
                              void* d_out, int out_size, void* d_ws, size_t ws_size,
                              hipStream_t stream) {
  const float* x_raw      = (const float*)d_in[1];
  const float* enc_w1     = (const float*)d_in[3];
  const float* enc_b1     = (const float*)d_in[4];
  const float* enc_w2     = (const float*)d_in[5];
  const float* enc_b2     = (const float*)d_in[6];
  const float* log_dt     = (const float*)d_in[7];
  const float* log_A_real = (const float*)d_in[8];
  const float* A_imag     = (const float*)d_in[9];
  const float* C_ri       = (const float*)d_in[10];
  const float* Dp         = (const float*)d_in[11];
  const float* out_w      = (const float*)d_in[12];
  const float* out_b      = (const float*)d_in[13];
  const float* norm_w     = (const float*)d_in[14];
  const float* norm_b     = (const float*)d_in[15];
  const float* wout_w     = (const float*)d_in[16];
  const float* wout_b     = (const float*)d_in[17];

  // workspace layout
  float* X    = (float*)d_ws;                        // BB*LMAX*HDIM f32
  float* Y    = X + XY_ELEMS;                        // BB*LMAX*HDIM f32
  float* SBuf = Y + XY_ELEMS;                        // scan chunk states
  bf16*  Abf  = (bf16*)(SBuf + SBUF_ELEMS);          // BB*LMAX*HDIM bf16
  bf16*  Wenc = Abf + XY_ELEMS;                      // 768*768 (enc_w2^T)
  bf16*  Wlay = Wenc + (size_t)HDIM * HDIM;          // 6*1536*768 (out_w, already (d,h))
  bf16*  Wfin = Wlay + (size_t)NL * 2 * HDIM * HDIM; // 48*768 (wout_w^T padded)

  // 1) weight conversion
  {
    int n = HDIM * HDIM;
    k_transpose_bf16<<<(n + 255) / 256, 256, 0, stream>>>(enc_w2, Wenc, HDIM, HDIM);
    int nl = NL * 2 * HDIM * HDIM;
    k_convert_bf16<<<(nl + 255) / 256, 256, 0, stream>>>(out_w, Wlay, nl);
    int nf = NOUTP * HDIM;
    k_build_wout<<<(nf + 255) / 256, 256, 0, stream>>>(wout_w, Wfin);
  }

  // 2) encoder
  {
    int M = BB * LMAX;                                    // 16384 tokens
    k_encoder1<<<(M * HDIM + 255) / 256, 256, 0, stream>>>(x_raw, enc_w1, enc_b1, Abf, M);
    int tm2 = M / 32, tn = HDIM / 16;
    k_gemm_bias<<<(tm2 * tn) / 4, 128, 0, stream>>>(Abf, Wenc, enc_b2, X,
                                                    HDIM, HDIM, tm2, tn, HDIM);
  }

  // 3) S4 layers
  int L = LMAX;
  int scan_blocks = (BB * PCH * HDIM) / 256;             // 384
  int comb_blocks = (BB * HDIM * NST) / 256;             // 384
  for (int i = 0; i < NL; ++i) {
    int ds = (i <= 2) ? 2 : 1;
    int Lk = L / ds;
    int Mk = BB * Lk;
    const float* ldt = log_dt + (size_t)i * HDIM;
    const float* lar = log_A_real + (size_t)i * HDIM * NST;
    const float* aim = A_imag + (size_t)i * HDIM * NST;
    const float* c0  = C_ri + ((size_t)i * 2 + 0) * HDIM * NST * 2;
    const float* c1  = C_ri + ((size_t)i * 2 + 1) * HDIM * NST * 2;
    const float* dv  = Dp + (size_t)i * HDIM;
    const bf16*  wl  = Wlay + (size_t)i * 2 * HDIM * HDIM;
    const float* ob  = out_b + (size_t)i * 2 * HDIM;

    // causal scan: phase1 -> combine -> phase3 (writes yacc into Y)
    k_scan_p1<<<scan_blocks, 256, 0, stream>>>(X, SBuf, ldt, lar, aim, L, 0);
    k_scan_combine<<<comb_blocks, 256, 0, stream>>>(SBuf, ldt, lar, aim, L, 0);
    k_scan_p3_fwd<<<scan_blocks, 256, 0, stream>>>(X, Y, SBuf, ldt, lar, aim, c0, dv, L);
    // anticausal scan: phase1 -> combine -> phase3 (+GELU, compaction to bf16)
    k_scan_p1<<<scan_blocks, 256, 0, stream>>>(X, SBuf, ldt, lar, aim, L, 1);
    k_scan_combine<<<comb_blocks, 256, 0, stream>>>(SBuf, ldt, lar, aim, L, 1);
    k_scan_p3_bwd<<<scan_blocks, 256, 0, stream>>>(X, Y, Abf, SBuf, ldt, lar, aim, c1, L, ds);

    int tm8 = Mk / 128, tn = HDIM / 16;
    k_gemm_glu<<<tm8 * tn, 128, 0, stream>>>(Abf, wl, ob, X, Y,
                                             HDIM, Lk, ds, L, tm8, tn);
    k_layernorm<<<Mk, 256, 0, stream>>>(Y, X, norm_w + (size_t)i * HDIM,
                                        norm_b + (size_t)i * HDIM);
    L = Lk;
  }

  // 4) head: (BB*256, 768) @ (768, 38) + bias, N padded to 48, guarded stores
  {
    int M = BB * L;                                        // 2048
    k_convert_bf16<<<(M * HDIM + 255) / 256, 256, 0, stream>>>(X, Abf, M * HDIM);
    int tm2 = M / 32, tn = NOUTP / 16;
    k_gemm_bias<<<(tm2 * tn) / 4, 128, 0, stream>>>(Abf, Wfin, wout_b, (float*)d_out,
                                                    HDIM, NOUT, tm2, tn, NOUT);
  }
}
